// TorchBlock_86904368267355
// MI455X (gfx1250) — matrix operand order repaired and checked by reading, not compiled, not run on hardware
//
#include <hip/hip_runtime.h>
#include <hip/hip_bf16.h>

typedef __attribute__((ext_vector_type(16))) __bf16 v16bf;
typedef __attribute__((ext_vector_type(8)))  float  v8f;

// ---------- helpers ----------
static __device__ inline unsigned short f2bf(float f) {
    union { float f; unsigned u; } c; c.f = f;
    unsigned u = c.u;
    unsigned r = (u + 0x7FFFu + ((u >> 16) & 1u)) >> 16;   // RNE
    return (unsigned short)r;
}

// Build a v16bf fragment from two contiguous 16-byte chunks (ISA A/B layouts).
static __device__ inline v16bf load_frag2(const unsigned short* p0,
                                          const unsigned short* p1) {
    v16bf r;
    uint4* u = reinterpret_cast<uint4*>(&r);
    u[0] = *reinterpret_cast<const uint4*>(p0);
    u[1] = *reinterpret_cast<const uint4*>(p1);
    return r;
}

static __device__ inline v8f wmma_bf16(v16bf a, v16bf b, v8f c) {
    return __builtin_amdgcn_wmma_f32_16x16x32_bf16(
        false, a, false, b, (short)0, c, false, false);
}

// ---------- fp32 -> bf16 convert ----------
__global__ void cvt_kernel(const float* __restrict__ in,
                           unsigned short* __restrict__ out, int n) {
    int i = blockIdx.x * blockDim.x + threadIdx.x;
    if (i < n) out[i] = f2bf(in[i]);
}

// ---------- LayerNorm: fp32 row -> bf16 row (E = 1024, 256 threads) ----------
__global__ __launch_bounds__(256) void ln_kernel(const float* __restrict__ x,
                                                 unsigned short* __restrict__ out) {
    constexpr int EW = 1024;
    const int row = blockIdx.x, tid = threadIdx.x;
    const float* xr = x + (size_t)row * EW;
    float4 v = *reinterpret_cast<const float4*>(xr + tid * 4);
    float s = v.x + v.y + v.z + v.w;
    float q = v.x * v.x + v.y * v.y + v.z * v.z + v.w * v.w;
    #pragma unroll
    for (int off = 16; off > 0; off >>= 1) {
        s += __shfl_xor(s, off);
        q += __shfl_xor(q, off);
    }
    __shared__ float ss[8], sq[8];
    const int wid = tid >> 5, lane = tid & 31;
    if (lane == 0) { ss[wid] = s; sq[wid] = q; }
    __syncthreads();
    float ts = 0.f, tq = 0.f;
    #pragma unroll
    for (int i = 0; i < 8; ++i) { ts += ss[i]; tq += sq[i]; }
    const float mean = ts * (1.0f / EW);
    const float var  = tq * (1.0f / EW) - mean * mean;
    const float rstd = rsqrtf(var + 1e-5f);
    unsigned short* orow = out + (size_t)row * EW + tid * 4;
    orow[0] = f2bf((v.x - mean) * rstd);
    orow[1] = f2bf((v.y - mean) * rstd);
    orow[2] = f2bf((v.z - mean) * rstd);
    orow[3] = f2bf((v.w - mean) * rstd);
}

// ---------- WMMA GEMM: C[M,N] = A[M,K] @ W[N,K]^T (+bias,+relu,+resid) ----------
// Block = 128 threads = 4 waves (2x2). Wave tile 64x64 -> block tile 128x128.
// 16 WMMAs per 32-deep K-step against 16 b128 loads: 1 load per WMMA.
template <bool BIAS, bool RELU, bool RESID, bool OUTF32, bool OUTBF16>
__global__ __launch_bounds__(128) void gemm_bf16_kernel(
    const unsigned short* __restrict__ A, const unsigned short* __restrict__ W,
    const float* __restrict__ bias, const float* __restrict__ resid,
    float* __restrict__ outf, unsigned short* __restrict__ outh,
    int M, int N, int K) {
    const int lane = threadIdx.x & 31, wid = threadIdx.x >> 5;
    const int g = lane >> 4, ln = lane & 15;
    const int wm = wid >> 1, wn = wid & 1;
    const int m0 = blockIdx.y * 128 + wm * 64;
    const int n0 = blockIdx.x * 128 + wn * 64;

    v8f acc[4][4];
    #pragma unroll
    for (int i = 0; i < 4; ++i)
        #pragma unroll
        for (int j = 0; j < 4; ++j) acc[i][j] = (v8f){};

    for (int kb = 0; kb < K; kb += 32) {
        v16bf a[4];
        #pragma unroll
        for (int mt = 0; mt < 4; ++mt) {
            const unsigned short* ap =
                A + (size_t)(m0 + mt * 16 + ln) * K + kb + g * 8;
            a[mt] = load_frag2(ap, ap + 16);            // A: two 8-elem runs
        }
        #pragma unroll
        for (int nt = 0; nt < 4; ++nt) {
            const unsigned short* bp =
                W + (size_t)(n0 + nt * 16 + ln) * K + kb + g * 8;
            v16bf bm = load_frag2(bp, bp + 16);         // B: two 8-elem runs
            #pragma unroll
            for (int mt = 0; mt < 4; ++mt)
                acc[mt][nt] = wmma_bf16(a[mt], bm, acc[mt][nt]);
        }
    }

    #pragma unroll
    for (int mt = 0; mt < 4; ++mt) {
        #pragma unroll
        for (int nt = 0; nt < 4; ++nt) {
            const int n = n0 + nt * 16 + ln;
            const float bv = BIAS ? bias[n] : 0.0f;
            #pragma unroll
            for (int r = 0; r < 8; ++r) {
                const int m = m0 + mt * 16 + r + 8 * g;   // C-layout row
                const size_t idx = (size_t)m * N + n;
                float val = acc[mt][nt][r];
                if (BIAS)  val += bv;
                if (RELU)  val = fmaxf(val, 0.0f);
                if (RESID) val += resid[idx];
                if (OUTF32)  outf[idx] = val;
                if (OUTBF16) outh[idx] = f2bf(val);
            }
        }
    }
}

// ---------- Flash attention: one wave per 16-query tile, 32-key blocks ----------
__global__ __launch_bounds__(128) void attn_kernel(
    const unsigned short* __restrict__ qkv, unsigned short* __restrict__ obuf) {
    constexpr int S = 2048, E = 1024, TE = 3072;
    const int lane = threadIdx.x & 31, wid = threadIdx.x >> 5;
    const int g = lane >> 4, ln = lane & 15;
    const int qt = blockIdx.x * 4 + wid;
    const int q0 = qt * 16;
    const int bh = blockIdx.y;
    const int b = bh >> 4, h = bh & 15;

    __shared__ unsigned short p_lds[4][16 * 32];
    __shared__ unsigned short v_lds[4][32 * 64];
    unsigned short* pl = p_lds[wid];
    unsigned short* vl = v_lds[wid];

    const size_t baserow = (size_t)b * S;

    // Q fragments (A-layout), loaded once: K-dim = head dim 64 = 2 chunks of 32
    v16bf qa[2];
    {
        const unsigned short* qp = qkv + (baserow + q0 + ln) * TE + h * 64;
        #pragma unroll
        for (int c = 0; c < 2; ++c)
            qa[c] = load_frag2(qp + c * 32 + g * 8, qp + c * 32 + 16 + g * 8);
    }

    v8f o[4];
    #pragma unroll
    for (int i = 0; i < 4; ++i) o[i] = (v8f){};
    float mrow[8], lrow[8];
    #pragma unroll
    for (int r = 0; r < 8; ++r) { mrow[r] = -1e30f; lrow[r] = 0.0f; }

    // Per-lane LDS byte offset of this wave's V tile row (generic addr low 32
    // bits == LDS offset per the aperture layout).
    const unsigned vrow_lds =
        (unsigned)(unsigned long long)(vl + lane * 64);

    const int nblk = (q0 + 16 + 31) >> 5;   // ceil((q0+16)/32) causal blocks
    for (int kb2 = 0; kb2 < nblk; ++kb2) {
        const int kb = kb2 * 32;

        // Stage V rows into LDS with the CDNA5 async DMA path: lane L pulls
        // key (kb+L)'s 64 contiguous dims straight into LDS (no VGPR bounce).
        {
            const unsigned long long gbase = (unsigned long long)
                (qkv + (baserow + kb + lane) * TE + 2 * E + h * 64);
            #pragma unroll
            for (int i = 0; i < 8; ++i) {
                unsigned long long ga = gbase + (unsigned long long)(i * 16);
                unsigned la = vrow_lds + i * 16;
                asm volatile("global_load_async_to_lds_b128 %0, %1, off"
                             :: "v"(la), "v"(ga) : "memory");
            }
        }

        // Scores S = Q K^T for 2 key n-tiles (B-layout over head dim)
        v8f s[2];
        #pragma unroll
        for (int t = 0; t < 2; ++t) {
            const unsigned short* kp =
                qkv + (baserow + kb + t * 16 + ln) * TE + E + h * 64;
            v16bf k0 = load_frag2(kp + g * 8, kp + 16 + g * 8);
            v16bf k1 = load_frag2(kp + 32 + g * 8, kp + 32 + 16 + g * 8);
            v8f z = (v8f){};
            s[t] = wmma_bf16(qa[0], k0, z);
            s[t] = wmma_bf16(qa[1], k1, s[t]);
        }

        // Online softmax: scale, causal mask, row max/sum via 16-lane shuffles
        float alpha[8];
        #pragma unroll
        for (int r = 0; r < 8; ++r) {
            const int q = q0 + r + 8 * g;
            float v0 = s[0][r] * 0.125f;
            float v1 = s[1][r] * 0.125f;
            if (kb + ln > q)      v0 = -1e30f;
            if (kb + 16 + ln > q) v1 = -1e30f;
            float mb = fmaxf(v0, v1);
            #pragma unroll
            for (int off = 8; off > 0; off >>= 1)
                mb = fmaxf(mb, __shfl_xor(mb, off, 16));
            const float mn = fmaxf(mrow[r], mb);
            alpha[r] = __expf(mrow[r] - mn);
            mrow[r] = mn;
            const float p0 = __expf(v0 - mn), p1 = __expf(v1 - mn);
            s[0][r] = p0; s[1][r] = p1;
            float ps = p0 + p1;
            #pragma unroll
            for (int off = 8; off > 0; off >>= 1)
                ps += __shfl_xor(ps, off, 16);
            lrow[r] = lrow[r] * alpha[r] + ps;
        }

        // Rescale running output
        #pragma unroll
        for (int vt = 0; vt < 4; ++vt)
            #pragma unroll
            for (int r = 0; r < 8; ++r) o[vt][r] *= alpha[r];

        // P (C-layout) -> LDS as bf16 16x32 tile
        #pragma unroll
        for (int t = 0; t < 2; ++t)
            #pragma unroll
            for (int r = 0; r < 8; ++r)
                pl[(r + 8 * g) * 32 + t * 16 + ln] = f2bf(s[t][r]);

        asm volatile("s_wait_dscnt 0" ::: "memory");     // P-tile LDS RAW
        asm volatile("s_wait_asynccnt 0" ::: "memory");  // V-tile DMA complete

        // P as A-fragment (16x32, K = keys)
        v16bf pa = load_frag2(pl + ln * 32 + g * 16, pl + ln * 32 + g * 16 + 8);

        // O += P @ V : 4 output n-tiles over head dim
        #pragma unroll
        for (int vt = 0; vt < 4; ++vt) {
            v16bf vf;
            unsigned short* sp = reinterpret_cast<unsigned short*>(&vf);
            const int d = vt * 16 + ln;
            #pragma unroll
            for (int e = 0; e < 16; ++e)
                sp[e] = vl[(g * 16 + e) * 64 + d];     // B-layout: K = key
            o[vt] = wmma_bf16(pa, vf, o[vt]);
        }
    }

    // Normalize and store bf16 output rows
    #pragma unroll
    for (int vt = 0; vt < 4; ++vt)
        #pragma unroll
        for (int r = 0; r < 8; ++r) {
            const float val = o[vt][r] / lrow[r];
            obuf[(baserow + q0 + r + 8 * g) * E + h * 64 + vt * 16 + ln] =
                f2bf(val);
        }
}

// ---------- host ----------
extern "C" void kernel_launch(void* const* d_in, const int* in_sizes, int n_in,
                              void* d_out, int out_size, void* d_ws, size_t ws_size,
                              hipStream_t stream) {
    (void)in_sizes; (void)n_in; (void)out_size; (void)ws_size;
    constexpr int B = 2, S = 2048, E = 1024, R = B * S;

    const float* x      = (const float*)d_in[0];
    const float* w_in   = (const float*)d_in[1];   // [3E, E]
    const float* w_out  = (const float*)d_in[2];   // [E, E]
    const float* w_fc   = (const float*)d_in[3];   // [4E, E]
    const float* b_fc   = (const float*)d_in[4];   // [4E]
    const float* w_proj = (const float*)d_in[5];   // [E, 4E]
    const float* b_proj = (const float*)d_in[6];   // [E]
    float* out = (float*)d_out;

    char* ws = (char*)d_ws;
    size_t off = 0;
    auto alloc = [&](size_t bytes) {
        char* p = ws + off;
        off += (bytes + 255) & ~(size_t)255;
        return p;
    };
    unsigned short* win_h   = (unsigned short*)alloc((size_t)3 * E * E * 2);
    unsigned short* wout_h  = (unsigned short*)alloc((size_t)E * E * 2);
    unsigned short* wfc_h   = (unsigned short*)alloc((size_t)4 * E * E * 2);
    unsigned short* wproj_h = (unsigned short*)alloc((size_t)4 * E * E * 2);
    unsigned short* h_h     = (unsigned short*)alloc((size_t)R * E * 2);  // reused for h2
    unsigned short* qkv_h   = (unsigned short*)alloc((size_t)R * 3 * E * 2);
    unsigned short* o_h     = (unsigned short*)alloc((size_t)R * E * 2);
    float*          x2_f    = (float*)alloc((size_t)R * E * 4);
    unsigned short* m_h     = (unsigned short*)alloc((size_t)R * 4 * E * 2);

    // 1) weights -> bf16
    auto cvt = [&](const float* src, unsigned short* dst, int n) {
        cvt_kernel<<<(n + 255) / 256, 256, 0, stream>>>(src, dst, n);
    };
    cvt(w_in,   win_h,   3 * E * E);
    cvt(w_out,  wout_h,  E * E);
    cvt(w_fc,   wfc_h,   4 * E * E);
    cvt(w_proj, wproj_h, 4 * E * E);

    // 2) h = LN(x) in bf16
    ln_kernel<<<R, 256, 0, stream>>>(x, h_h);

    // 3) qkv = h @ w_in^T   [R, 3E] bf16
    gemm_bf16_kernel<false, false, false, false, true>
        <<<dim3(3 * E / 128, R / 128), 128, 0, stream>>>(
            h_h, win_h, nullptr, nullptr, nullptr, qkv_h, R, 3 * E, E);

    // 4) flash attention -> o bf16 [R, E]
    attn_kernel<<<dim3(S / 16 / 4, B * 16), 128, 0, stream>>>(qkv_h, o_h);

    // 5) x2 = o @ w_out^T + x   (fp32)
    gemm_bf16_kernel<false, false, true, true, false>
        <<<dim3(E / 128, R / 128), 128, 0, stream>>>(
            o_h, wout_h, nullptr, x, x2_f, nullptr, R, E, E);

    // 6) h2 = LN(x2) in bf16 (reuse h buffer)
    ln_kernel<<<R, 256, 0, stream>>>(x2_f, h_h);

    // 7) m = relu(h2 @ w_fc^T + b_fc)   [R, 4E] bf16
    gemm_bf16_kernel<true, true, false, false, true>
        <<<dim3(4 * E / 128, R / 128), 128, 0, stream>>>(
            h_h, wfc_h, b_fc, nullptr, nullptr, m_h, R, 4 * E, E);

    // 8) out = m @ w_proj^T + b_proj + x2   (fp32)
    gemm_bf16_kernel<true, false, true, true, false>
        <<<dim3(E / 128, R / 128), 128, 0, stream>>>(
            m_h, wproj_h, b_proj, x2_f, out, nullptr, R, E, 4 * E);
}
